// GatedDeltaNet_10479720202643
// MI455X (gfx1250) — compile-verified, matrix operand order checked by reading, not run on hardware
//
#include <hip/hip_runtime.h>
#include <hip/hip_bf16.h>
#include <stdint.h>

// ---- problem constants (match reference) ----
#define DMODEL 1024
#define NH     16
#define DKH    64
#define DVH    128
#define KDIM   1024      // NH*DKH
#define VDIM   2048      // NH*DVH
#define BB     2
#define TT     4096
#define NBT    (BB*TT)   // 8192 rows

typedef __attribute__((ext_vector_type(16))) __bf16 v16bf;
typedef __attribute__((ext_vector_type(8)))  float  v8f;

union FragBF { unsigned int u[8]; v16bf v; };

__device__ __forceinline__ unsigned int f2bfbits(float f) {
  unsigned int u = __float_as_uint(f);
  return (u + 0x7FFFu + ((u >> 16) & 1u)) >> 16;   // RNE f32->bf16
}
__device__ __forceinline__ float sigmoidf_(float x){ return 1.0f/(1.0f+__expf(-x)); }
__device__ __forceinline__ float siluf_(float x){ return x*sigmoidf_(x); }

// ---- convert f32 -> packed bf16 pairs (pairs contiguous along K, even K in low 16 bits) ----
__global__ void pack_x_bf16(const float* __restrict__ x, unsigned int* __restrict__ out, int npairs){
  int i = blockIdx.x*blockDim.x + threadIdx.x;
  if (i >= npairs) return;
  out[i] = f2bfbits(x[2*i]) | (f2bfbits(x[2*i+1]) << 16);
}

// W[K,N] f32 row-major -> fragment-major tiled bf16 pairs:
//   out dword index i = ((s*N + n)*2 + half)*8 + j
//   with kp = s*16 + 4*half + j + (j>=4 ? 4 : 0), pair = {bf16(W[2kp,n]), bf16(W[2kp+1,n])}
// This makes each lane's 8-dword B fragment contiguous (2x b128, 16B aligned).
__global__ void pack_w_bf16_tiled(const float* __restrict__ W, unsigned int* __restrict__ out,
                                  int Kp, int N){
  int i = blockIdx.x*blockDim.x + threadIdx.x;
  if (i >= Kp*N) return;
  int j    = i & 7;
  int half = (i >> 3) & 1;
  int n    = (i >> 4) % N;
  int s    = i / (N*16);
  int kp   = s*16 + 4*half + j + ((j >= 4) ? 4 : 0);
  out[i] = f2bfbits(W[(size_t)(2*kp)*N + n]) | (f2bfbits(W[(size_t)(2*kp+1)*N + n]) << 16);
}

// ---- bf16 WMMA GEMM: C[M,N] f32 = A[M,K] x B[K,N]
// Ap = [M, Kp] dwords (bf16 pairs along K), Bt = tiled layout from pack_w_bf16_tiled, Kp = K/2.
// WG = 256 thr (8 waves); WG tile 128x128; each wave computes 32x64 (2x4 WMMA tiles).
// All A/B traffic is global_load_b128 (index-form addressing keeps global addrspace).
__global__ __launch_bounds__(256)
void wmma_gemm_bf16(const unsigned int* __restrict__ Ap,
                    const unsigned int* __restrict__ Bt,
                    float* __restrict__ C, int M, int N, int Kp){
  const int lane = threadIdx.x & 31;
  const int wave = threadIdx.x >> 5;
  const int half = lane >> 4;        // selects K sub-range per 16-bit A/B layout
  const int l16  = lane & 15;
  const int m0 = blockIdx.y*128 + (wave & 3)*32;
  const int n0 = blockIdx.x*128 + (wave >> 2)*64;
  v8f acc[2][4];
#pragma unroll
  for (int i = 0; i < 2; ++i)
#pragma unroll
    for (int c = 0; c < 4; ++c) acc[i][c] = (v8f){};
  const size_t aOff0 = (size_t)(m0 + l16)*Kp;
  const size_t aOff1 = (size_t)(m0 + 16 + l16)*Kp;
  // per-lane contiguous B fragment offsets (advance by 16*N dwords per k-slab)
  size_t bOff[4];
#pragma unroll
  for (int c = 0; c < 4; ++c)
    bOff[c] = (((size_t)(n0 + c*16 + l16))*2 + half)*8;
  const size_t bStep = (size_t)N*16;
  for (int kp0 = 0; kp0 < Kp; kp0 += 16){
    if (kp0 + 16 < Kp){                // global_prefetch_b8 for next k-slab
      __builtin_prefetch(Ap + aOff0 + kp0 + 16, 0, 0);
      __builtin_prefetch(Ap + aOff1 + kp0 + 16, 0, 0);
      __builtin_prefetch(Bt + bOff[0] + bStep, 0, 0);
      __builtin_prefetch(Bt + bOff[2] + bStep, 0, 0);
    }
    FragBF a0, a1, b[4];
#pragma unroll
    for (int j = 0; j < 8; ++j){
      // 16-bit A 16x32 layout: VGPR j<4 -> K=8*half+2j ; j>=4 -> K=16+8*half+2(j-4)  (pair index kp)
      const int kp = kp0 + j + 4*half + ((j >= 4) ? 4 : 0);
      a0.u[j] = Ap[aOff0 + kp];
      a1.u[j] = Ap[aOff1 + kp];
    }
#pragma unroll
    for (int c = 0; c < 4; ++c){
#pragma unroll
      for (int j = 0; j < 8; ++j) b[c].u[j] = Bt[bOff[c] + j];   // 2x global_load_b128 each
      bOff[c] += bStep;
    }
#pragma unroll
    for (int c = 0; c < 4; ++c){
      acc[0][c] = __builtin_amdgcn_wmma_f32_16x16x32_bf16(false, a0.v, false, b[c].v, (short)0, acc[0][c], false, false);
      acc[1][c] = __builtin_amdgcn_wmma_f32_16x16x32_bf16(false, a1.v, false, b[c].v, (short)0, acc[1][c], false, false);
    }
  }
  // C/D layout: lane n = n0+l16 ; VGPR r -> m = r + 8*half
#pragma unroll
  for (int i = 0; i < 2; ++i){
#pragma unroll
    for (int c = 0; c < 4; ++c){
#pragma unroll
      for (int r = 0; r < 8; ++r){
        const int mr = m0 + i*16 + r + 8*half;
        C[(size_t)mr*N + n0 + c*16 + l16] = acc[i][c][r];
      }
    }
  }
}

// ---- g = -exp(A_log)*softplus(x@w_a + dt_bias) ; beta = 2*sigmoid(x@w_b) ----
__global__ void gates_kernel(const float* __restrict__ x, const float* __restrict__ wa,
                             const float* __restrict__ wb, const float* __restrict__ Alog,
                             const float* __restrict__ dtb,
                             float* __restrict__ g, float* __restrict__ beta){
  int i = blockIdx.x*blockDim.x + threadIdx.x;       // i = row*NH + h
  if (i >= NBT*NH) return;
  int h = i & (NH-1);
  int row = i >> 4;
  const float* xr = x + (size_t)row*DMODEL;
  float a = 0.f, b = 0.f;
  for (int d = 0; d < DMODEL; ++d){
    float xv = xr[d];
    a += xv * wa[d*NH + h];
    b += xv * wb[d*NH + h];
  }
  float s = a + dtb[h];
  float sp = (s > 20.f) ? s : __logf(1.f + __expf(s));
  g[i]    = -__expf(Alog[h]) * sp;
  beta[i] = 2.f * sigmoidf_(b);
}

// ---- depthwise causal conv1d (K=4) + SiLU ----
__global__ void conv_silu_kernel(const float* __restrict__ in, const float* __restrict__ w,
                                 float* __restrict__ out, int C, int total){
  int i = blockIdx.x*blockDim.x + threadIdx.x;
  if (i >= total) return;
  int c  = i % C;
  int bt = i / C;
  int t  = bt % TT;
  float acc = 0.f;
#pragma unroll
  for (int j = 0; j < 4; ++j){
    int tt = t - 3 + j;
    if (tt >= 0) acc += in[(size_t)(bt - t + tt)*C + c] * w[c*4 + j];
  }
  out[i] = siluf_(acc);
}

// ---- per-head l2 normalization over DK=64 (in place) ----
__global__ void l2norm_kernel(float* __restrict__ p, int nvec){
  int i = blockIdx.x*blockDim.x + threadIdx.x;       // vec = row*NH + h ; contiguous 64-elem groups
  if (i >= nvec) return;
  float* v = p + (size_t)i*DKH;
  float buf[DKH];
  float s = 1e-6f;
#pragma unroll
  for (int j = 0; j < DKH; ++j){ buf[j] = v[j]; s += buf[j]*buf[j]; }
  float r = rsqrtf(s);
#pragma unroll
  for (int j = 0; j < DKH; ++j) v[j] = buf[j]*r;
}

// ---- gated delta-rule scan: one WG per (b,h); thread j owns state column S[:,j].
// State held as float4 S4[16] (64 VGPRs). k_t/q_t double-buffered in LDS so each step
// needs a single barrier; LDS reads are float4 (ds_load_b128), k cached in registers.
__global__ __launch_bounds__(128)
void scan_kernel(const float* __restrict__ qc, const float* __restrict__ kc,
                 const float* __restrict__ vc, const float* __restrict__ g,
                 const float* __restrict__ beta, float* __restrict__ o){
  const int bh = blockIdx.x;                  // 0..31
  const int b  = bh >> 4, h = bh & 15;
  const int j  = threadIdx.x;                 // 0..127 (DV column)
  __shared__ __align__(16) float sk[2][DKH];
  __shared__ __align__(16) float sq[2][DKH];
  float4 S4[16];
#pragma unroll
  for (int i = 0; i < 16; ++i) S4[i] = (float4){0.f, 0.f, 0.f, 0.f};
  const float scale = 0.125f;                 // DK^-0.5
  const size_t kqBase = (size_t)(b*TT)*KDIM + h*DKH;
  // preload t=0 into buffer 0
  if (j < DKH){
    sk[0][j] = kc[kqBase + j];
    sq[0][j] = qc[kqBase + j];
  }
  for (int t = 0; t < TT; ++t){
    __syncthreads();                          // buf[t&1] ready; prev reads of buf[(t+1)&1] done
    const int cur = t & 1;
    if (j < DKH && (t + 1) < TT){             // stage next step while computing this one
      const size_t nb = kqBase + (size_t)(t + 1)*KDIM;
      sk[cur ^ 1][j] = kc[nb + j];
      sq[cur ^ 1][j] = qc[nb + j];
    }
    const int row = b*TT + t;
    const float eg = __expf(g[row*NH + h]);
    const float bt = beta[row*NH + h];
    const float vj = vc[(size_t)row*VDIM + h*DVH + j];
    const float4* k4 = (const float4*)sk[cur];
    const float4* q4 = (const float4*)sq[cur];
    float4 kk[16];
#pragma unroll
    for (int i = 0; i < 16; ++i) kk[i] = k4[i];   // 16x ds_load_b128, cached for both uses
    float mem = 0.f;
#pragma unroll
    for (int i = 0; i < 16; ++i){                 // readout on pre-decay state
      mem += S4[i].x*kk[i].x + S4[i].y*kk[i].y + S4[i].z*kk[i].z + S4[i].w*kk[i].w;
    }
    mem *= eg;                                    // fold decay into readout
    const float delta = (vj - mem) * bt;
    float acc = 0.f;
#pragma unroll
    for (int i = 0; i < 16; ++i){                 // decayed rank-1 update + output dot
      const float4 q = q4[i];
      S4[i].x = S4[i].x*eg + kk[i].x*delta;
      S4[i].y = S4[i].y*eg + kk[i].y*delta;
      S4[i].z = S4[i].z*eg + kk[i].z*delta;
      S4[i].w = S4[i].w*eg + kk[i].w*delta;
      acc += S4[i].x*q.x + S4[i].y*q.y + S4[i].z*q.z + S4[i].w*q.w;
    }
    o[(size_t)row*VDIM + h*DVH + j] = acc * scale;
  }
}

// ---- gated RMSNorm over DV=128, emit bf16 row-major (ready as WMMA A operand) ----
__global__ void rmsnorm_gate_kernel(const float* __restrict__ o, const float* __restrict__ gate,
                                    const float* __restrict__ nw, unsigned short* __restrict__ yb){
  int i = blockIdx.x*blockDim.x + threadIdx.x;       // row*NH + h
  if (i >= NBT*NH) return;
  size_t base = (size_t)(i >> 4)*VDIM + (size_t)(i & 15)*DVH;
  float ss = 0.f;
  for (int j = 0; j < DVH; ++j){ float v = o[base+j]; ss += v*v; }
  float r = rsqrtf(ss * (1.0f/DVH) + 1e-5f);
  for (int j = 0; j < DVH; ++j){
    float v = o[base+j]*r*nw[j]*siluf_(gate[base+j]);
    yb[base+j] = (unsigned short)f2bfbits(v);
  }
}

extern "C" void kernel_launch(void* const* d_in, const int* in_sizes, int n_in,
                              void* d_out, int out_size, void* d_ws, size_t ws_size,
                              hipStream_t stream){
  const float* x       = (const float*)d_in[0];
  const float* w_q     = (const float*)d_in[1];
  const float* w_k     = (const float*)d_in[2];
  const float* w_v     = (const float*)d_in[3];
  const float* w_a     = (const float*)d_in[4];
  const float* w_b     = (const float*)d_in[5];
  const float* w_g     = (const float*)d_in[6];
  const float* w_out   = (const float*)d_in[7];
  const float* A_log   = (const float*)d_in[8];
  const float* dt_bias = (const float*)d_in[9];
  const float* conv_q  = (const float*)d_in[10];
  const float* conv_k  = (const float*)d_in[11];
  const float* conv_v  = (const float*)d_in[12];
  const float* norm_w  = (const float*)d_in[13];

  char* ws = (char*)d_ws;
  size_t off = 0;
  auto alloc = [&](size_t bytes) -> char* {
    char* p = ws + off; off += (bytes + 255) & ~(size_t)255; return p;
  };
  unsigned int* XB  = (unsigned int*)alloc((size_t)NBT*DMODEL*2);  // x as bf16 pairs
  unsigned int* WQP = (unsigned int*)alloc((size_t)DMODEL*KDIM*2);
  unsigned int* WKP = (unsigned int*)alloc((size_t)DMODEL*KDIM*2);
  unsigned int* WVP = (unsigned int*)alloc((size_t)DMODEL*VDIM*2);
  unsigned int* WGP = (unsigned int*)alloc((size_t)DMODEL*VDIM*2);
  unsigned int* WOP = (unsigned int*)alloc((size_t)VDIM*DMODEL*2);
  float* Q    = (float*)alloc((size_t)NBT*KDIM*4);
  float* K_   = (float*)alloc((size_t)NBT*KDIM*4);
  float* V    = (float*)alloc((size_t)NBT*VDIM*4);
  float* GATE = (float*)alloc((size_t)NBT*VDIM*4);
  float* QC   = (float*)alloc((size_t)NBT*KDIM*4);
  float* KC   = (float*)alloc((size_t)NBT*KDIM*4);
  float* VC   = (float*)alloc((size_t)NBT*VDIM*4);
  float* Gg   = (float*)alloc((size_t)NBT*NH*4);
  float* BETA = (float*)alloc((size_t)NBT*NH*4);
  // dead-buffer reuse: scan output O overlays Q+K (exactly 64MB contiguous);
  // bf16 y overlays the first half of V.
  float*          O  = Q;
  unsigned short* YB = (unsigned short*)V;

  const int TPB = 256;
  // 1) pack activations + weights to bf16 (weights in fragment-major tiled layout)
  { int np = NBT*DMODEL/2;
    pack_x_bf16<<<(np+TPB-1)/TPB, TPB, 0, stream>>>(x, XB, np); }
  { int n = (DMODEL/2)*KDIM;
    pack_w_bf16_tiled<<<(n+TPB-1)/TPB, TPB, 0, stream>>>(w_q, WQP, DMODEL/2, KDIM);
    pack_w_bf16_tiled<<<(n+TPB-1)/TPB, TPB, 0, stream>>>(w_k, WKP, DMODEL/2, KDIM); }
  { int n = (DMODEL/2)*VDIM;
    pack_w_bf16_tiled<<<(n+TPB-1)/TPB, TPB, 0, stream>>>(w_v, WVP, DMODEL/2, VDIM);
    pack_w_bf16_tiled<<<(n+TPB-1)/TPB, TPB, 0, stream>>>(w_g, WGP, DMODEL/2, VDIM); }
  { int n = (VDIM/2)*DMODEL;
    pack_w_bf16_tiled<<<(n+TPB-1)/TPB, TPB, 0, stream>>>(w_out, WOP, VDIM/2, DMODEL); }

  // 2) projection GEMMs (WMMA bf16 -> f32); WG tile 128x128
  { dim3 grd(KDIM/128, NBT/128);
    wmma_gemm_bf16<<<grd, 256, 0, stream>>>(XB, WQP, Q,  NBT, KDIM, DMODEL/2);
    wmma_gemm_bf16<<<grd, 256, 0, stream>>>(XB, WKP, K_, NBT, KDIM, DMODEL/2); }
  { dim3 grd(VDIM/128, NBT/128);
    wmma_gemm_bf16<<<grd, 256, 0, stream>>>(XB, WVP, V,    NBT, VDIM, DMODEL/2);
    wmma_gemm_bf16<<<grd, 256, 0, stream>>>(XB, WGP, GATE, NBT, VDIM, DMODEL/2); }

  // 3) decay/beta gates (tiny GEMV fused with nonlinearity)
  { int n = NBT*NH;
    gates_kernel<<<(n+TPB-1)/TPB, TPB, 0, stream>>>(x, w_a, w_b, A_log, dt_bias, Gg, BETA); }

  // 4) causal conv + SiLU
  { int n = NBT*KDIM;
    conv_silu_kernel<<<(n+TPB-1)/TPB, TPB, 0, stream>>>(Q,  conv_q, QC, KDIM, n);
    conv_silu_kernel<<<(n+TPB-1)/TPB, TPB, 0, stream>>>(K_, conv_k, KC, KDIM, n); }
  { int n = NBT*VDIM;
    conv_silu_kernel<<<(n+TPB-1)/TPB, TPB, 0, stream>>>(V,  conv_v, VC, VDIM, n); }

  // 5) per-head l2norm on q,k
  { int n = NBT*NH;
    l2norm_kernel<<<(n+TPB-1)/TPB, TPB, 0, stream>>>(QC, n);
    l2norm_kernel<<<(n+TPB-1)/TPB, TPB, 0, stream>>>(KC, n); }

  // 6) gated delta-rule recurrence: 32 WGs (one per b,h), 128 threads each
  scan_kernel<<<BB*NH, 128, 0, stream>>>(QC, KC, VC, Gg, BETA, O);

  // 7) gated RMSNorm -> bf16
  { int n = NBT*NH;
    rmsnorm_gate_kernel<<<(n+TPB-1)/TPB, TPB, 0, stream>>>(O, GATE, norm_w, YB); }

  // 8) output projection GEMM -> d_out (f32)
  { dim3 grd(DMODEL/128, NBT/128);
    wmma_gemm_bf16<<<grd, 256, 0, stream>>>((const unsigned int*)YB, WOP, (float*)d_out,
                                            NBT, DMODEL, VDIM/2); }
}
